// HeuristicGCN_13726715478604
// MI455X (gfx1250) — matrix operand order repaired and checked by reading, not compiled, not run on hardware
//
#include <hip/hip_runtime.h>
#include <stdint.h>

// ---------- types ----------
typedef __attribute__((ext_vector_type(16))) __bf16 v16bf;
typedef __attribute__((ext_vector_type(8)))  float  v8f;

__device__ __forceinline__ unsigned short f2bf_u(float f) {
  union { float f; uint32_t u; } v; v.f = f;
  uint32_t u = v.u;
  uint32_t r = u + 0x7FFFu + ((u >> 16) & 1u);   // round-to-nearest-even
  return (unsigned short)(r >> 16);
}
__device__ __forceinline__ uint32_t pack2(float lo, float hi) {
  return (uint32_t)f2bf_u(lo) | ((uint32_t)f2bf_u(hi) << 16);
}
__device__ __forceinline__ __bf16 f2bf(float f) {
  union { unsigned short u; __bf16 b; } v; v.u = f2bf_u(f); return v.b;
}

// Fragment loaders: 16-byte vector LDS loads (lower to ds_load_b128).
// A layout (16-bit 16x32): lane's 16 halfs live at [base+0..7] and [base+16..23].
union FragU { struct { uint4 a, b; } q; v16bf v; };
__device__ __forceinline__ v16bf ld_fragA(const unsigned short* base) {
  FragU u;
  u.q.a = *(const uint4*)(base);
  u.q.b = *(const uint4*)(base + 16);
  return u.v;
}
// B layout with column-major LDS tile: lane's 16 halfs live at [base+0..7] and [base+16..23].
__device__ __forceinline__ v16bf ld_fragB(const unsigned short* base) {
  FragU u;
  u.q.a = *(const uint4*)(base);
  u.q.b = *(const uint4*)(base + 16);
  return u.v;
}

// ---------- small elementwise / graph kernels ----------
__global__ void bins_kernel(const int* __restrict__ degrees, int* __restrict__ bins, int n) {
  int i = blockIdx.x * 256 + threadIdx.x;
  if (i < n) {
    int b = (int)log1pf((float)degrees[i]);        // trunc like astype(int32)
    b = b < 0 ? 0 : (b > 99 ? 99 : b);
    bins[i] = b;
  }
}

__global__ void count_kernel(const int* __restrict__ col, int* __restrict__ deg, int e) {
  int i = blockIdx.x * 256 + threadIdx.x;
  if (i < e) atomicAdd(&deg[col[i]], 1);
}

__global__ void dinv_kernel(const int* __restrict__ deg, float* __restrict__ dinv, int n) {
  int i = blockIdx.x * 256 + threadIdx.x;
  if (i < n) {
    int d = deg[i];
    dinv[i] = (d > 0) ? rsqrtf((float)d) : 0.0f;
  }
}

// agg[col] += h[row] * (dinv[row]*dinv[col]); one thread handles 4 channels of one edge
__global__ void scatter_kernel(const float* __restrict__ h, const int* __restrict__ row,
                               const int* __restrict__ col, const float* __restrict__ dinv,
                               float* __restrict__ agg, int E) {
  int i = blockIdx.x * 256 + threadIdx.x;
  int e = i >> 6;
  int c = (i & 63) << 2;
  if (e < E) {
    int r = row[e], cl = col[e];
    float nrm = dinv[r] * dinv[cl];
    float4 hv = *(const float4*)(h + (size_t)r * 256 + c);
    float* dst = agg + (size_t)cl * 256 + c;
    atomicAdd(dst + 0, hv.x * nrm);
    atomicAdd(dst + 1, hv.y * nrm);
    atomicAdd(dst + 2, hv.z * nrm);
    atomicAdd(dst + 3, hv.w * nrm);
  }
}

__global__ void bias_relu_kernel(const float* __restrict__ agg, const float* __restrict__ b,
                                 float* __restrict__ z, int n) {
  int i = blockIdx.x * 256 + threadIdx.x;
  if (i < n) z[i] = fmaxf(agg[i] + b[i & 255], 0.0f);
}

// ---------- tiled WMMA bf16 GEMM ----------
// C[M x N] = A[M x K] * B[K x N] (+bias, +relu). Block tile 64x128, 256 thr = 8 waves,
// each wave: one 16-row tile x four 16-col tiles (A-frag reused across 4 WMMAs).
// MODE 0: A = f32 matrix (lda). MODE 1: A = concat(node_emb, degree_emb[bins]).
#define AST 40     // halfs per A row
#define BCST 40    // halfs per B column (column-major tile)

template<int MODE>
__device__ __forceinline__ float loadA(const float* __restrict__ A, int lda,
                                       const float* __restrict__ nodeEmb,
                                       const float* __restrict__ degEmb,
                                       const int* __restrict__ bins,
                                       int gr, int k, int M, int Kreal) {
  if (gr >= M || k >= Kreal) return 0.0f;
  if (MODE == 0) return A[(size_t)gr * lda + k];
  return (k < 256) ? nodeEmb[(size_t)gr * 256 + k] : degEmb[bins[gr] * 16 + (k - 256)];
}

template<int MODE>
__global__ __launch_bounds__(256)
void gemm_kernel(const float* __restrict__ A, int lda,
                 const float* __restrict__ nodeEmb, const float* __restrict__ degEmb,
                 const int* __restrict__ bins,
                 const float* __restrict__ B, const float* __restrict__ bias,
                 float* __restrict__ C,
                 int M, int Kreal, int Kchunks, int N, int doRelu) {
  __shared__ unsigned short As[64 * AST];     // 5.0 KB
  __shared__ unsigned short Bs[128 * BCST];   // 10.0 KB
  int tid  = threadIdx.x;
  int lane = tid & 31;
  int wave = tid >> 5;
  int row0 = blockIdx.x * 64;
  int col0 = blockIdx.y * 128;
  int mt   = wave >> 1;          // M tile 0..3
  int nt0  = (wave & 1) * 4;     // N tile base 0 or 4

  // staging coordinates
  int ar  = tid >> 2;            // A: row 0..63
  int acg = (tid & 3) * 8;       // A: 8 cols
  int brp = (tid >> 4) * 2;      // B: k-row pair 0..30
  int bcg = (tid & 15) * 8;      // B: 8 cols

  v8f acc[4] = {};

  for (int kc = 0; kc < Kchunks; ++kc) {
    int k0 = kc * 32;

    // prefetch next K-chunk (global_prefetch_b8)
    if (kc + 1 < Kchunks) {
      int kn = k0 + 32;
      if (MODE == 0) {
        if (row0 + ar < M) __builtin_prefetch(&A[(size_t)(row0 + ar) * lda + kn + acg], 0, 0);
      } else {
        if (row0 + ar < M && kn + acg < 256)
          __builtin_prefetch(&nodeEmb[(size_t)(row0 + ar) * 256 + kn + acg], 0, 0);
      }
      if (kn + brp < Kreal) __builtin_prefetch(&B[(size_t)(kn + brp) * N + col0 + bcg], 0, 0);
    }

    { // stage A: 64x32, packed bf16, one ds_store_b128 per thread
      int gr = row0 + ar;
      uint4 q;
      uint32_t w[4];
      #pragma unroll
      for (int c = 0; c < 4; ++c) {
        int k = k0 + acg + c * 2;
        float v0 = loadA<MODE>(A, lda, nodeEmb, degEmb, bins, gr, k, M, Kreal);
        float v1 = loadA<MODE>(A, lda, nodeEmb, degEmb, bins, gr, k + 1, M, Kreal);
        w[c] = pack2(v0, v1);
      }
      q.x = w[0]; q.y = w[1]; q.z = w[2]; q.w = w[3];
      *(uint4*)&As[ar * AST + acg] = q;
    }
    { // stage B: 32x128 -> column-major bf16, packed k-pairs (ds_store_b32)
      int gk0 = k0 + brp, gk1 = k0 + brp + 1;
      #pragma unroll
      for (int c = 0; c < 8; ++c) {
        float v0 = (gk0 < Kreal) ? B[(size_t)gk0 * N + col0 + bcg + c] : 0.0f;
        float v1 = (gk1 < Kreal) ? B[(size_t)gk1 * N + col0 + bcg + c] : 0.0f;
        *(uint32_t*)&Bs[(bcg + c) * BCST + brp] = pack2(v0, v1);
      }
    }
    __syncthreads();

    v16bf a = ld_fragA(&As[((mt << 4) + (lane & 15)) * AST + ((lane >> 4) << 3)]);
    #pragma unroll
    for (int t = 0; t < 4; ++t) {
      v16bf b = ld_fragB(&Bs[(((nt0 + t) << 4) + (lane & 15)) * BCST + ((lane >> 4) << 3)]);
      acc[t] = __builtin_amdgcn_wmma_f32_16x16x32_bf16(false, a, false, b, (short)0, acc[t], false, false);
    }
    __syncthreads();
  }

  // store D: VGPR r -> M = r (+8 for hi lanes), N = lane&15
  int mrow = row0 + (mt << 4) + ((lane >> 4) << 3);
  int ncol = lane & 15;
  #pragma unroll
  for (int r = 0; r < 8; ++r) {
    int gm = mrow + r;
    if (gm < M) {
      #pragma unroll
      for (int t = 0; t < 4; ++t) {
        int gn = col0 + ((nt0 + t) << 4) + ncol;
        float v = acc[t][r];
        if (bias) v += bias[gn];
        if (doRelu) v = fmaxf(v, 0.0f);
        C[(size_t)gm * N + gn] = v;
      }
    }
  }
}

// ---------- decoder: 16 edges per block, full MLP fused ----------
// x = [z[a]*z[b] (256) | heur emb (64)] -> relu(x W1 + b1) -> relu(. W2 + b2) -> . W3 + b3
#define XS_ST 328   // halfs; 656 B row stride (16B multiple)
#define W1C_ST 40   // W1 chunk column-major: 40 halfs per column
#define X2_ST 264   // halfs; 528 B row stride (16B multiple)
#define W2C_ST 264  // W2 column-major: 264 halfs per column (256 + pad)
__global__ __launch_bounds__(256)
void decode_kernel(const float* __restrict__ z, const int* __restrict__ edge,
                   const int* __restrict__ heur, const float* __restrict__ heurEmb,
                   const float* __restrict__ W1, const float* __restrict__ b1,
                   const float* __restrict__ W2, const float* __restrict__ b2,
                   const float* __restrict__ W3, const float* __restrict__ b3,
                   float* __restrict__ out) {
  __shared__ unsigned short xs[16 * XS_ST];     // 16 x 320 bf16  (10.3 KB)
  __shared__ unsigned short Bsc[256 * W1C_ST];  // W1 chunk col-major (20 KB)
  __shared__ unsigned short x2s[16 * X2_ST];    // 16 x 256 bf16  (8.3 KB)
  __shared__ unsigned short W2c[64 * W2C_ST];   // W2 col-major   (33 KB)
  __shared__ float x3s[16 * 68];                // 16 x 64 f32    (4.3 KB)
  int tid  = threadIdx.x;
  int lane = tid & 31;
  int wave = tid >> 5;
  int e0   = blockIdx.x * 16;

  // stage all of W2 (256x64) column-major, packed k-pairs
  for (int i = tid; i < 64 * 128; i += 256) {
    int cc = i & 63, rp = (i >> 6) * 2;
    *(uint32_t*)&W2c[cc * W2C_ST + rp] =
        pack2(W2[(size_t)rp * 64 + cc], W2[(size_t)(rp + 1) * 64 + cc]);
  }

  // build x tile (pairs of k)
  for (int i = tid; i < 16 * 160; i += 256) {
    int m = i / 160;
    int k = (i - m * 160) * 2;
    int e = e0 + m;
    int na = edge[e * 2 + 0], nb = edge[e * 2 + 1];
    float v0, v1;
    if (k < 256) {
      v0 = z[(size_t)na * 256 + k] * z[(size_t)nb * 256 + k];
      v1 = z[(size_t)na * 256 + k + 1] * z[(size_t)nb * 256 + k + 1];
    } else {
      int kk = k - 256;
      int hi = kk >> 4, hj = kk & 15;
      int idx = heur[e * 4 + hi];
      const float* he = &heurEmb[(hi * 100 + idx) * 16 + hj];
      v0 = he[0]; v1 = he[1];
    }
    *(uint32_t*)&xs[m * XS_ST + k] = pack2(v0, v1);
  }
  __syncthreads();

  // W1 GEMM: 16x320 @ 320x256; wave w owns cols [32w, 32w+32)
  v8f c0 = {}, c1 = {};
  for (int kc = 0; kc < 10; ++kc) {
    int k0 = kc * 32;
    if (kc < 9)
      __builtin_prefetch(&W1[(size_t)(k0 + 32 + (tid >> 3)) * 256 + (tid & 7) * 32], 0, 0);
    for (int i = tid; i < 256 * 16; i += 256) {
      int cc = i & 255, rp = (i >> 8) * 2;
      *(uint32_t*)&Bsc[cc * W1C_ST + rp] =
          pack2(W1[(size_t)(k0 + rp) * 256 + cc], W1[(size_t)(k0 + rp + 1) * 256 + cc]);
    }
    __syncthreads();
    v16bf a = ld_fragA(&xs[(lane & 15) * XS_ST + k0 + ((lane >> 4) << 3)]);
    #pragma unroll
    for (int t = 0; t < 2; ++t) {
      v16bf b = ld_fragB(&Bsc[(wave * 32 + (t << 4) + (lane & 15)) * W1C_ST + ((lane >> 4) << 3)]);
      if (t == 0) c0 = __builtin_amdgcn_wmma_f32_16x16x32_bf16(false, a, false, b, (short)0, c0, false, false);
      else        c1 = __builtin_amdgcn_wmma_f32_16x16x32_bf16(false, a, false, b, (short)0, c1, false, false);
    }
    __syncthreads();
  }
  { // bias + relu -> x2 (bf16 in LDS)
    int mb = (lane >> 4) << 3;
    int nn = lane & 15;
    #pragma unroll
    for (int r = 0; r < 8; ++r) {
      #pragma unroll
      for (int t = 0; t < 2; ++t) {
        int col = wave * 32 + t * 16 + nn;
        float v = ((t == 0) ? c0[r] : c1[r]) + b1[col];
        x2s[(mb + r) * X2_ST + col] = f2bf_u(fmaxf(v, 0.0f));
      }
    }
  }
  __syncthreads();

  // W2 GEMM: 16x256 @ 256x64; waves 0..3 own one 16-col tile each
  if (wave < 4) {
    v8f c = {};
    #pragma unroll
    for (int kc = 0; kc < 8; ++kc) {
      int k0 = kc * 32;
      v16bf a = ld_fragA(&x2s[(lane & 15) * X2_ST + k0 + ((lane >> 4) << 3)]);
      v16bf b = ld_fragB(&W2c[(wave * 16 + (lane & 15)) * W2C_ST + k0 + ((lane >> 4) << 3)]);
      c = __builtin_amdgcn_wmma_f32_16x16x32_bf16(false, a, false, b, (short)0, c, false, false);
    }
    int mb = (lane >> 4) << 3;
    int nn = lane & 15;
    #pragma unroll
    for (int r = 0; r < 8; ++r) {
      int col = wave * 16 + nn;
      x3s[(mb + r) * 68 + col] = fmaxf(c[r] + b2[col], 0.0f);
    }
  }
  __syncthreads();

  if (tid < 16) {
    float s = b3[0];
    #pragma unroll
    for (int j = 0; j < 64; ++j) s += x3s[tid * 68 + j] * W3[j];
    out[e0 + tid] = s;
  }
}

// ---------- launcher ----------
extern "C" void kernel_launch(void* const* d_in, const int* in_sizes, int n_in,
                              void* d_out, int out_size, void* d_ws, size_t ws_size,
                              hipStream_t stream) {
  const int*   edge_index = (const int*)d_in[0];   // (2, 300000)
  const int*   pos_edge   = (const int*)d_in[1];   // (200000, 2)
  const int*   neg_edge   = (const int*)d_in[2];
  const int*   pos_h      = (const int*)d_in[3];   // (200000, 4)
  const int*   neg_h      = (const int*)d_in[4];
  const int*   degrees    = (const int*)d_in[5];   // (100000,)
  const float* node_emb   = (const float*)d_in[6]; // (100000, 256)
  const float* degree_emb = (const float*)d_in[7]; // (100, 16)
  const float* proj_W     = (const float*)d_in[8]; // (272, 256)
  const float* proj_b     = (const float*)d_in[9];
  const float* conv_W     = (const float*)d_in[10]; // (2, 256, 256)
  const float* conv_b     = (const float*)d_in[11]; // (2, 256)
  const float* heur_emb   = (const float*)d_in[12]; // (4, 100, 16)
  const float* W1 = (const float*)d_in[13];
  const float* b1 = (const float*)d_in[14];
  const float* W2 = (const float*)d_in[15];
  const float* b2 = (const float*)d_in[16];
  const float* W3 = (const float*)d_in[17];
  const float* b3 = (const float*)d_in[18];
  float* out = (float*)d_out;

  const int N = 100000, E = 300000, EC = 200000, HID = 256;

  char* ws = (char*)d_ws;
  size_t off = 0;
  auto alloc = [&](size_t bytes) -> void* {
    void* p = ws + off;
    off = (off + bytes + 255) & ~(size_t)255;
    return p;
  };
  float* z    = (float*)alloc((size_t)N * HID * 4);
  float* h    = (float*)alloc((size_t)N * HID * 4);
  float* agg  = (float*)alloc((size_t)N * HID * 4);
  int*   bins = (int*)alloc((size_t)N * 4);
  int*   deg  = (int*)alloc((size_t)N * 4);
  float* dinv = (float*)alloc((size_t)N * 4);

  hipMemsetAsync(deg, 0, (size_t)N * 4, stream);
  bins_kernel<<<(N + 255) / 256, 256, 0, stream>>>(degrees, bins, N);
  count_kernel<<<(E + 255) / 256, 256, 0, stream>>>(edge_index + E, deg, E);
  dinv_kernel<<<(N + 255) / 256, 256, 0, stream>>>(deg, dinv, N);

  dim3 gGemm((N + 63) / 64, HID / 128);
  // proj: z = concat(node_emb, degree_emb[bins]) @ proj_W + proj_b
  gemm_kernel<1><<<gGemm, 256, 0, stream>>>(nullptr, 0, node_emb, degree_emb, bins,
                                            proj_W, proj_b, z, N, 272, 9, HID, 0);

  for (int l = 0; l < 2; ++l) {
    gemm_kernel<0><<<gGemm, 256, 0, stream>>>(z, HID, nullptr, nullptr, nullptr,
                                              conv_W + (size_t)l * HID * HID, nullptr, h,
                                              N, HID, 8, HID, 0);
    hipMemsetAsync(agg, 0, (size_t)N * HID * 4, stream);
    scatter_kernel<<<((size_t)E * 64 + 255) / 256, 256, 0, stream>>>(
        h, edge_index, edge_index + E, dinv, agg, E);
    bias_relu_kernel<<<((size_t)N * HID + 255) / 256, 256, 0, stream>>>(
        agg, conv_b + l * HID, z, N * HID);
  }

  decode_kernel<<<EC / 16, 256, 0, stream>>>(z, pos_edge, pos_h, heur_emb,
                                             W1, b1, W2, b2, W3, b3, out);
  decode_kernel<<<EC / 16, 256, 0, stream>>>(z, neg_edge, neg_h, heur_emb,
                                             W1, b1, W2, b2, W3, b3, out + EC);
}
